// cmil_mnist_9947144257679
// MI455X (gfx1250) — matrix operand order repaired and checked by reading, not compiled, not run on hardware
//
#include <hip/hip_runtime.h>
#include <hip/hip_bf16.h>
#include <math.h>

// ---------------------------------------------------------------------------
// MI455X (gfx1250) implementation. Heavy GEMM-like work (conv2/conv3 implicit
// GEMM, FC mu/logvar) runs on v_wmma_f32_16x16x32_f16; everything small runs
// on VALU with LDS reductions. wave32 layouts per CDNA5 ISA 7.12.2.
// ---------------------------------------------------------------------------

typedef __attribute__((ext_vector_type(16))) _Float16 v16h;
typedef __attribute__((ext_vector_type(8)))  _Float16 v8h;
typedef __attribute__((ext_vector_type(8)))  float    v8f;

union HvU { v16h v; v8h h[2]; };

#define N_INST 8192
#define N_BAGS 128
#define LATENT 64

__device__ __forceinline__ v8f wmma_f16(v16h a, v16h b, v8f c) {
  // (neg_a, A, neg_b, B, c_mod, C, reuse_a, reuse_b)
  return __builtin_amdgcn_wmma_f32_16x16x32_f16(false, a, false, b, (short)0, c,
                                                false, false);
}

// ---------------------------------------------------------------------------
// Prep: convert conv2/conv3 weights to f16 [cout][k] (already that layout),
// and build combined FC weight Wth[128][512] f16 = [W_mu^T ; W_lv^T].
// ---------------------------------------------------------------------------
__global__ void prep_kernel(const float* __restrict__ w2, const float* __restrict__ w3,
                            const float* __restrict__ Wmu, const float* __restrict__ Wlv,
                            _Float16* __restrict__ w2h, _Float16* __restrict__ w3h,
                            _Float16* __restrict__ Wth)
{
  int i = blockIdx.x * blockDim.x + threadIdx.x;
  if (i < 9216) { w2h[i] = (_Float16)w2[i]; w3h[i] = (_Float16)w3[i]; }
  if (i < 128 * 512) {
    int n = i >> 9, k = i & 511;
    float v = (n < 64) ? Wmu[k * 64 + n] : Wlv[k * 64 + (n - 64)];
    Wth[i] = (_Float16)v;
  }
}

__global__ void init_kernel(float* __restrict__ out) {
  if (threadIdx.x < 3) out[threadIdx.x] = 0.0f;
}

// ---------------------------------------------------------------------------
// Fused encoder: one block per image (128 thr = 4 waves).
//   conv1 (VALU, K=27) -> conv2 (WMMA, M=49->64, K=288, N=32)
//   -> conv3 (WMMA, M=16, K=288, N=32) -> H row [512] f32 + f16.
// ---------------------------------------------------------------------------
__global__ __launch_bounds__(128) void enc_kernel(
    const float* __restrict__ bag,
    const float* __restrict__ w1, const float* __restrict__ b1,
    const float* __restrict__ b2, const float* __restrict__ b3,
    const _Float16* __restrict__ w2h, const _Float16* __restrict__ w3h,
    float* __restrict__ H, _Float16* __restrict__ Hh)
{
  __shared__ __align__(16) char smem[58432];
  _Float16* out1Pad = (_Float16*)smem;            // [32][16][16]  16384 B
  _Float16* patch2  = (_Float16*)(smem + 16384);  // [64][288]     36864 B
  _Float16* imgPad  = (_Float16*)(smem + 16384);  // [3][30][30]   aliases patch2
  _Float16* out2Pad = (_Float16*)(smem + 53248);  // [32][9][9]     5184 B
  _Float16* patch3  = (_Float16*)(smem + 16384);  // [16][288]     aliases patch2

  const int img  = blockIdx.x;
  const int tid  = threadIdx.x;
  const int lane = tid & 31;
  const int wv   = tid >> 5;
  const int l15  = lane & 15;
  const int halfA = (lane >= 16) ? 8 : 0;    // A-fragment K base per ISA layout
  const int halfB = (lane >= 16) ? 16 : 0;   // unused; B loads use halfA (same k order as A)
  const float* src = bag + (size_t)img * 2352;

  if (img + 1 < N_INST) __builtin_prefetch(src + 2352, 0, 1);

  // phase 1: padded f16 image into LDS
  for (int i = tid; i < 2700; i += 128) imgPad[i] = (_Float16)0.0f;
  for (int i = tid; i < 8192; i += 128) out1Pad[i] = (_Float16)0.0f;
  __syncthreads();
  for (int i = tid; i < 2352; i += 128) {
    int ci = i / 784, r = i % 784, y = r / 28, x = r % 28;
    imgPad[ci * 900 + (y + 1) * 30 + (x + 1)] = (_Float16)src[i];
  }
  __syncthreads();

  // phase 2: conv1 3->32, s2 p1: 28 -> 14 (direct, K=27)
  for (int idx = tid; idx < 6272; idx += 128) {
    int c = idx / 196, r = idx % 196, y = r / 14, x = r % 14;
    float acc = b1[c];
    for (int ci = 0; ci < 3; ci++)
      for (int ky = 0; ky < 3; ky++)
        for (int kx = 0; kx < 3; kx++)
          acc += (float)imgPad[ci * 900 + (2 * y + ky) * 30 + (2 * x + kx)] *
                 w1[((c * 3 + ci) * 3 + ky) * 3 + kx];
    out1Pad[c * 256 + (y + 1) * 16 + (x + 1)] = (_Float16)fmaxf(acc, 0.0f);
  }
  __syncthreads();

  // phase 3: im2col patch for conv2 (M=64 padded, K=288), zero conv2 out pad
  for (int i = tid; i < 64 * 288; i += 128) {
    int m = i / 288, k = i % 288;
    _Float16 val = (_Float16)0.0f;
    if (m < 49) {
      int oy = m / 7, ox = m % 7;
      int ci = k / 9, t = k % 9, dy = t / 3, dx = t % 3;
      val = out1Pad[ci * 256 + (2 * oy + dy) * 16 + (2 * ox + dx)];
    }
    patch2[i] = val;
  }
  for (int i = tid; i < 2592; i += 128) out2Pad[i] = (_Float16)0.0f;
  __syncthreads();

  // conv2 WMMA: 4 Mtiles x 2 Ntiles = 8 jobs over 4 waves
  for (int job = wv; job < 8; job += 4) {
    int mtile = job >> 1, ntile = job & 1;
    int mrow = mtile * 16 + l15;
    int ncol = ntile * 16 + l15;
    const _Float16* arow = patch2 + mrow * 288;
    const _Float16* brow = w2h + ncol * 288;
    v8f c = {};
    for (int k0 = 0; k0 < 288; k0 += 32) {
      HvU a, b;
      a.h[0] = *(const v8h*)(arow + k0 + halfA);
      a.h[1] = *(const v8h*)(arow + k0 + 16 + halfA);
      b.h[0] = *(const v8h*)(brow + k0 + halfA);
      b.h[1] = *(const v8h*)(brow + k0 + 16 + halfA);
      c = wmma_f16(a.v, b.v, c);
    }
    int mbase = mtile * 16 + ((lane >= 16) ? 8 : 0);
    float bias = b2[ncol];
    for (int vv = 0; vv < 8; vv++) {
      int m = mbase + vv;
      if (m < 49) {
        float r = fmaxf(c[vv] + bias, 0.0f);
        out2Pad[ncol * 81 + (m / 7 + 1) * 9 + (m % 7 + 1)] = (_Float16)r;
      }
    }
  }
  __syncthreads();

  // phase 4: im2col patch for conv3 (M=16, K=288)
  for (int i = tid; i < 16 * 288; i += 128) {
    int m = i / 288, k = i % 288;
    int oy = m >> 2, ox = m & 3;
    int ci = k / 9, t = k % 9, dy = t / 3, dx = t % 3;
    patch3[i] = out2Pad[ci * 81 + (2 * oy + dy) * 9 + (2 * ox + dx)];
  }
  __syncthreads();

  // conv3 WMMA: 1 Mtile x 2 Ntiles on waves 0,1; H row epilogue
  if (wv < 2) {
    int ncol = wv * 16 + l15;
    const _Float16* arow = patch3 + l15 * 288;
    const _Float16* brow = w3h + ncol * 288;
    v8f c = {};
    for (int k0 = 0; k0 < 288; k0 += 32) {
      HvU a, b;
      a.h[0] = *(const v8h*)(arow + k0 + halfA);
      a.h[1] = *(const v8h*)(arow + k0 + 16 + halfA);
      b.h[0] = *(const v8h*)(brow + k0 + halfA);
      b.h[1] = *(const v8h*)(brow + k0 + 16 + halfA);
      c = wmma_f16(a.v, b.v, c);
    }
    int mbase = (lane >= 16) ? 8 : 0;
    float bias = b3[ncol];
    float*    Hrow  = H  + (size_t)img * 512;
    _Float16* Hhrow = Hh + (size_t)img * 512;
    for (int vv = 0; vv < 8; vv++) {
      int m = mbase + vv;                  // spatial s = y*4+x
      float r = fmaxf(c[vv] + bias, 0.0f);
      Hrow[ncol * 16 + m]  = r;            // H[n, c*16 + y*4 + x]
      Hhrow[ncol * 16 + m] = (_Float16)r;
    }
  }
}

// ---------------------------------------------------------------------------
// FC WMMA GEMM: [8192,512] @ [512,128] (mu||lv) + fused rsample epilogue.
// One M-tile per block; wave w handles latent cols 16w..16w+15 (mu and lv).
// ---------------------------------------------------------------------------
__global__ __launch_bounds__(128) void fc_kernel(
    const _Float16* __restrict__ Hh, const _Float16* __restrict__ Wth,
    const float* __restrict__ bmu, const float* __restrict__ blv,
    const float* __restrict__ eps,
    float* __restrict__ mu, float* __restrict__ stdv, float* __restrict__ z)
{
  const int tid = threadIdx.x;
  const int lane = tid & 31;
  const int wv = tid >> 5;
  const int l15 = lane & 15;
  const int halfA = (lane >= 16) ? 8 : 0;
  const int halfB = (lane >= 16) ? 16 : 0;
  const int m0 = blockIdx.x * 16;
  const int mrow = m0 + l15;
  const int n = wv * 16 + l15;  // latent column 0..63
  const _Float16* arow  = Hh + (size_t)mrow * 512;
  const _Float16* brow0 = Wth + (size_t)n * 512;
  const _Float16* brow1 = Wth + (size_t)(n + 64) * 512;
  v8f cmu = {}, clv = {};
  for (int k0 = 0; k0 < 512; k0 += 32) {
    HvU a, bm_, bl_;
    a.h[0]   = *(const v8h*)(arow + k0 + halfA);
    a.h[1]   = *(const v8h*)(arow + k0 + 16 + halfA);
    bm_.h[0] = *(const v8h*)(brow0 + k0 + halfA);
    bm_.h[1] = *(const v8h*)(brow0 + k0 + 16 + halfA);
    bl_.h[0] = *(const v8h*)(brow1 + k0 + halfA);
    bl_.h[1] = *(const v8h*)(brow1 + k0 + 16 + halfA);
    cmu = wmma_f16(a.v, bm_.v, cmu);
    clv = wmma_f16(a.v, bl_.v, clv);
  }
  const int mbase = m0 + ((lane >= 16) ? 8 : 0);
  float bmv = bmu[n], blvv = blv[n];
  for (int vv = 0; vv < 8; vv++) {
    int m = mbase + vv;
    float muv = cmu[vv] + bmv;
    float lvv = clv[vv] + blvv;
    float sd  = expf(0.5f * lvv);
    size_t o  = (size_t)m * 64 + n;
    float zz  = muv + sd * eps[o];
    mu[o] = muv; stdv[o] = sd; z[o] = zz;
  }
}

// score[m] = z[m] . w_fc + b_fc
__global__ void score_kernel(const float* __restrict__ z, const float* __restrict__ wfc,
                             const float* __restrict__ bfc, float* __restrict__ score)
{
  int m = blockIdx.x * blockDim.x + threadIdx.x;
  if (m < N_INST) {
    float s = bfc[0];
    for (int n = 0; n < 64; n++) s += z[(size_t)m * 64 + n] * wfc[n];
    score[m] = s;
  }
}

// segment bookkeeping: counts + exclusive starts (bag_idx sorted 0..B-1)
__global__ void seg_kernel(const int* __restrict__ bidx,
                           int* __restrict__ starts, int* __restrict__ cnts)
{
  __shared__ int c[N_BAGS];
  int tid = threadIdx.x;
  for (int i = tid; i < N_BAGS; i += 256) c[i] = 0;
  __syncthreads();
  for (int i = tid; i < N_INST; i += 256) atomicAdd(&c[bidx[i]], 1);
  __syncthreads();
  if (tid == 0) {
    int s = 0;
    for (int b = 0; b < N_BAGS; b++) { starts[b] = s; cnts[b] = c[b]; s += c[b]; }
  }
}

// ---------------------------------------------------------------------------
// Per-bag: bag_enc mean -> pmu/plv/pstd, seg_max + first-argmax (replicating
// the reference's "local index used globally" bug), BCE aux, KL terms.
// ---------------------------------------------------------------------------
__global__ __launch_bounds__(256) void bag_kernel(
    const float* __restrict__ H,
    const float* __restrict__ Wdsm, const float* __restrict__ bdsm,
    const float* __restrict__ Wdsv, const float* __restrict__ bdsv,
    const float* __restrict__ score, const float* __restrict__ blab,
    const int* __restrict__ starts, const int* __restrict__ cnts,
    const float* __restrict__ mu, const float* __restrict__ stdv,
    const float* __restrict__ z,
    int* __restrict__ sel, float* __restrict__ outAcc)
{
  __shared__ float redA[256];
  __shared__ float redB[256];
  __shared__ int   redI[256];
  __shared__ float stat[4];
  __shared__ int   sel_s;
  int b = blockIdx.x, tid = threadIdx.x;
  int start = starts[b], cnt = cnts[b];
  float fc = (float)cnt;

  float pm = 0.f, pv = 0.f;
  for (int c = tid; c < 512; c += 256) {
    float ssum = 0.f;
    for (int i = 0; i < cnt; i++) ssum += H[(size_t)(start + i) * 512 + c];
    ssum /= fc;
    pm += ssum * Wdsm[c];
    pv += ssum * Wdsv[c];
  }
  redA[tid] = pm; redB[tid] = pv;
  __syncthreads();
  for (int off = 128; off > 0; off >>= 1) {
    if (tid < off) { redA[tid] += redA[tid + off]; redB[tid] += redB[tid + off]; }
    __syncthreads();
  }
  if (tid == 0) {
    float pmu_ = redA[0] + bdsm[0];
    float plv_ = redB[0] + bdsv[0];
    stat[0] = pmu_; stat[2] = plv_; stat[1] = expf(0.5f * plv_);
  }
  __syncthreads();

  // segment max
  float vmax = -1e30f;
  for (int i = tid; i < cnt; i += 256) vmax = fmaxf(vmax, score[start + i]);
  redA[tid] = vmax; __syncthreads();
  for (int off = 128; off > 0; off >>= 1) {
    if (tid < off) redA[tid] = fmaxf(redA[tid], redA[tid + off]);
    __syncthreads();
  }
  float smax = redA[0];
  __syncthreads();
  // first (min) within-bag index achieving the max
  int cand = 0x7fffffff;
  for (int i = tid; i < cnt; i += 256) if (score[start + i] == smax) cand = min(cand, i);
  redI[tid] = cand; __syncthreads();
  for (int off = 128; off > 0; off >>= 1) {
    if (tid < off) redI[tid] = min(redI[tid], redI[tid + off]);
    __syncthreads();
  }
  if (tid == 0) {
    int local = redI[0];          // used as a GLOBAL row index (source bug)
    sel_s = local; sel[b] = local;
    float y = blab[b];
    float lsp = (smax > 0.f) ? -log1pf(expf(-smax)) : (smax - log1pf(expf(smax)));
    float lsn = (smax < 0.f) ? -log1pf(expf(smax))  : (-smax - log1pf(expf(-smax)));
    float auxb = -(y * lsp + (1.f - y) * lsn);
    atomicAdd(&outAcc[2], auxb / (float)N_BAGS);
  }
  __syncthreads();

  float pmu_ = stat[0], pstd_ = stat[1], plv_ = stat[2];
  int s = sel_s;
  float term = 0.f;
  if (tid < LATENT) {
    float zh = z[(size_t)s * 64 + tid];
    float mh = mu[(size_t)s * 64 + tid];
    float sh = stdv[(size_t)s * 64 + tid];
    float dq = zh - mh;
    float lq = -0.5f * (2.f * logf(sh) + dq * dq / (sh * sh));
    float dp = zh - pmu_;
    float lp = -0.5f * (2.f * logf(pstd_) + dp * dp / (pstd_ * pstd_));
    term = lq - lp;
  }
  redA[tid] = term; __syncthreads();
  for (int off = 128; off > 0; off >>= 1) {
    if (tid < off) redA[tid] += redA[tid + off];
    __syncthreads();
  }
  if (tid == 0) {
    float kl = 0.5f * (pmu_ * pmu_ + pstd_ * pstd_ - plv_ - 1.f) + redA[0];
    atomicAdd(&outAcc[1], kl / (float)N_BAGS);
  }
}

// ---------------------------------------------------------------------------
// Decoder on the 128 selected instances (tiny): FC -> 3x ConvT -> sigmoid,
// MSE against x_t[sel].
// ---------------------------------------------------------------------------
__global__ __launch_bounds__(256) void dec_kernel(
    const float* __restrict__ bag, const float* __restrict__ z,
    const float* __restrict__ Wdec, const float* __restrict__ bdec,
    const float* __restrict__ dw1, const float* __restrict__ db1,
    const float* __restrict__ dw2, const float* __restrict__ db2,
    const float* __restrict__ dw3, const float* __restrict__ db3,
    const int* __restrict__ sel, float* __restrict__ outAcc)
{
  __shared__ float A0[512];        // [32][4][4]
  __shared__ float A1[32 * 49];    // [32][7][7]
  __shared__ float A2[32 * 196];   // [32][14][14]
  __shared__ float red[256];
  int b = blockIdx.x, tid = threadIdx.x;
  int s = sel[b];
  const float* zr = z + (size_t)s * 64;

  for (int c = tid; c < 512; c += 256) {
    float acc = bdec[c];
    for (int l = 0; l < 64; l++) acc += zr[l] * Wdec[l * 512 + c];
    A0[c] = fmaxf(acc, 0.0f);
  }
  __syncthreads();

  // convT1: [32,4,4] -> [32,7,7]   (oy = 2*iy + ky - 1)
  for (int idx = tid; idx < 32 * 49; idx += 256) {
    int o = idx / 49, r = idx % 49, oy = r / 7, ox = r % 7;
    float acc = db1[o];
    for (int ky = 0; ky < 3; ky++) {
      int ty = oy + 1 - ky;
      if (ty < 0 || (ty & 1)) continue;
      int iy = ty >> 1; if (iy >= 4) continue;
      for (int kx = 0; kx < 3; kx++) {
        int tx = ox + 1 - kx;
        if (tx < 0 || (tx & 1)) continue;
        int ix = tx >> 1; if (ix >= 4) continue;
        for (int i = 0; i < 32; i++)
          acc += A0[i * 16 + iy * 4 + ix] * dw1[((i * 32 + o) * 3 + ky) * 3 + kx];
      }
    }
    A1[idx] = fmaxf(acc, 0.0f);
  }
  __syncthreads();

  // convT2: [32,7,7] -> [32,14,14]
  for (int idx = tid; idx < 32 * 196; idx += 256) {
    int o = idx / 196, r = idx % 196, oy = r / 14, ox = r % 14;
    float acc = db2[o];
    for (int ky = 0; ky < 3; ky++) {
      int ty = oy + 1 - ky;
      if (ty < 0 || (ty & 1)) continue;
      int iy = ty >> 1; if (iy >= 7) continue;
      for (int kx = 0; kx < 3; kx++) {
        int tx = ox + 1 - kx;
        if (tx < 0 || (tx & 1)) continue;
        int ix = tx >> 1; if (ix >= 7) continue;
        for (int i = 0; i < 32; i++)
          acc += A1[i * 49 + iy * 7 + ix] * dw2[((i * 32 + o) * 3 + ky) * 3 + kx];
      }
    }
    A2[idx] = fmaxf(acc, 0.0f);
  }
  __syncthreads();

  // convT3 + sigmoid + squared error vs bag[sel] flat
  const float* xm = bag + (size_t)s * 2352;
  float part = 0.0f;
  for (int idx = tid; idx < 3 * 784; idx += 256) {
    int o = idx / 784, r = idx % 784, oy = r / 28, ox = r % 28;
    float acc = db3[o];
    for (int ky = 0; ky < 3; ky++) {
      int ty = oy + 1 - ky;
      if (ty < 0 || (ty & 1)) continue;
      int iy = ty >> 1; if (iy >= 14) continue;
      for (int kx = 0; kx < 3; kx++) {
        int tx = ox + 1 - kx;
        if (tx < 0 || (tx & 1)) continue;
        int ix = tx >> 1; if (ix >= 14) continue;
        for (int i = 0; i < 32; i++)
          acc += A2[i * 196 + iy * 14 + ix] * dw3[((i * 3 + o) * 3 + ky) * 3 + kx];
      }
    }
    float xr = 1.0f / (1.0f + expf(-acc));
    float d = xr - xm[idx];
    part += d * d;
  }
  red[tid] = part; __syncthreads();
  for (int off = 128; off > 0; off >>= 1) {
    if (tid < off) red[tid] += red[tid + off];
    __syncthreads();
  }
  if (tid == 0) atomicAdd(&outAcc[0], red[0] / (float)(N_BAGS * 2352));
}

// ---------------------------------------------------------------------------
extern "C" void kernel_launch(void* const* d_in, const int* in_sizes, int n_in,
                              void* d_out, int out_size, void* d_ws, size_t ws_size,
                              hipStream_t stream)
{
  (void)in_sizes; (void)n_in; (void)out_size; (void)ws_size;
  const float* bag  = (const float*)d_in[0];
  const int*   bidx = (const int*)d_in[1];
  const float* blab = (const float*)d_in[2];
  const float* eps  = (const float*)d_in[3];
  const float* w1   = (const float*)d_in[4];  const float* b1   = (const float*)d_in[5];
  const float* w2   = (const float*)d_in[6];  const float* b2   = (const float*)d_in[7];
  const float* w3   = (const float*)d_in[8];  const float* b3   = (const float*)d_in[9];
  const float* Wmu  = (const float*)d_in[10]; const float* bmu  = (const float*)d_in[11];
  const float* Wlv  = (const float*)d_in[12]; const float* blv  = (const float*)d_in[13];
  const float* Wdsm = (const float*)d_in[14]; const float* bdsm = (const float*)d_in[15];
  const float* Wdsv = (const float*)d_in[16]; const float* bdsv = (const float*)d_in[17];
  const float* wfc  = (const float*)d_in[18]; const float* bfc  = (const float*)d_in[19];
  const float* Wdec = (const float*)d_in[20]; const float* bdec = (const float*)d_in[21];
  const float* dw1  = (const float*)d_in[22]; const float* db1  = (const float*)d_in[23];
  const float* dw2  = (const float*)d_in[24]; const float* db2  = (const float*)d_in[25];
  const float* dw3  = (const float*)d_in[26]; const float* db3  = (const float*)d_in[27];

  char* ws = (char*)d_ws;
  float*    H     = (float*)(ws + 0);              // 8192*512*4  = 16777216
  _Float16* Hh    = (_Float16*)(ws + 16777216);    // 8192*512*2  =  8388608
  float*    mu    = (float*)(ws + 25165824);       // 8192*64*4   =  2097152
  float*    stdv  = (float*)(ws + 27262976);
  float*    z     = (float*)(ws + 29360128);
  float*    score = (float*)(ws + 31457280);       // 8192*4
  _Float16* w2h   = (_Float16*)(ws + 31490048);    // 9216*2
  _Float16* w3h   = (_Float16*)(ws + 31508480);    // 9216*2
  _Float16* Wth   = (_Float16*)(ws + 31526912);    // 128*512*2
  int*      starts= (int*)(ws + 31657984);
  int*      cnts  = (int*)(ws + 31658496);
  int*      sel   = (int*)(ws + 31659008);

  float* out = (float*)d_out;  // [recon, kl, aux]

  init_kernel<<<dim3(1), dim3(32), 0, stream>>>(out);
  prep_kernel<<<dim3(256), dim3(256), 0, stream>>>(w2, w3, Wmu, Wlv, w2h, w3h, Wth);
  enc_kernel<<<dim3(N_INST), dim3(128), 0, stream>>>(bag, w1, b1, b2, b3, w2h, w3h, H, Hh);
  fc_kernel<<<dim3(N_INST / 16), dim3(128), 0, stream>>>(Hh, Wth, bmu, blv, eps, mu, stdv, z);
  score_kernel<<<dim3(32), dim3(256), 0, stream>>>(z, wfc, bfc, score);
  seg_kernel<<<dim3(1), dim3(256), 0, stream>>>(bidx, starts, cnts);
  bag_kernel<<<dim3(N_BAGS), dim3(256), 0, stream>>>(H, Wdsm, bdsm, Wdsv, bdsv, score,
                                                     blab, starts, cnts, mu, stdv, z,
                                                     sel, out);
  dec_kernel<<<dim3(N_BAGS), dim3(256), 0, stream>>>(bag, z, Wdec, bdec, dw1, db1,
                                                     dw2, db2, dw3, db3, sel, out);
}
